// CrossScaleInjection_82308753260594
// MI455X (gfx1250) — compile-verified, matrix operand order checked by reading, not run on hardware
//
#include <hip/hip_runtime.h>
#include <hip/hip_bf16.h>

#define B_     4
#define ND     2048
#define NC     8192
#define DMODEL 256
#define DMICRO 11
#define TOPKN  32

typedef __attribute__((ext_vector_type(16))) __bf16 v16bf;
typedef __attribute__((ext_vector_type(8)))  float  v8f;

// Load a 16x32 bf16 WMMA fragment (A layout; B via [col][k]-major weights or
// row-major K^T) from a row-major tile with leading dim `ld` (elements).
// Lane l<16 : row l,    K = 0..7  and 16..23
// Lane l>=16: row l-16, K = 8..15 and 24..31
__device__ __forceinline__ v16bf load_frag(const __bf16* base, int ld, int lane) {
  const int r  = lane & 15;
  const int k0 = (lane >> 4) << 3;
  const __bf16* p = base + r * ld + k0;
  union { uint4 u[2]; v16bf v; } x;
  x.u[0] = *(const uint4*)(p);
  x.u[1] = *(const uint4*)(p + 16);
  return x.v;
}

// ---------------------------------------------------------------------------
// One-time weight transpose+convert: Wt[c][k] = (bf16) W[k][c].
// LDS-tiled 32x32 so both the read and the write are coalesced.
// ---------------------------------------------------------------------------
__global__ __launch_bounds__(256)
void transpose_w(const float* __restrict__ W, __bf16* __restrict__ Wt)
{
  __shared__ float tile[32][33];
  const int tx = threadIdx.x & 31, ty = threadIdx.x >> 5;   // 8 rows/pass
  const int kb = (blockIdx.x & 7) * 32;                     // k base
  const int cb = (blockIdx.x >> 3) * 32;                    // col base
  #pragma unroll
  for (int i = ty; i < 32; i += 8)
    tile[i][tx] = W[(size_t)(kb + i) * DMODEL + cb + tx];
  __syncthreads();
  #pragma unroll
  for (int i = ty; i < 32; i += 8)
    Wt[(size_t)(cb + i) * DMODEL + kb + tx] = (__bf16)tile[tx][i];
}

// ---------------------------------------------------------------------------
// x_micro = micro @ mp_w + mp_b        (K=11, pure VALU; 16 rows per block)
// ---------------------------------------------------------------------------
__global__ __launch_bounds__(256)
void micro_proj(const float* __restrict__ micro, const float* __restrict__ mp_w,
                const float* __restrict__ mp_b, float* __restrict__ xm)
{
  const int rowBase = blockIdx.x * 16;
  const int tid = threadIdx.x;                    // = output column (256)
  __shared__ float sM[16 * DMICRO];
  if (tid < 16 * DMICRO) sM[tid] = micro[rowBase * DMICRO + tid];
  __syncthreads();
  float wcol[DMICRO];
  #pragma unroll
  for (int j = 0; j < DMICRO; ++j) wcol[j] = mp_w[j * DMODEL + tid];
  const float bia = mp_b[tid];
  #pragma unroll
  for (int i = 0; i < 16; ++i) {
    float v = bia;
    #pragma unroll
    for (int j = 0; j < DMICRO; ++j) v += sM[i * DMICRO + j] * wcol[j];
    xm[(size_t)(rowBase + i) * DMODEL + tid] = v;
  }
}

// ---------------------------------------------------------------------------
// GEMM vs pre-transposed bf16 weight: out = alpha*(A@W) [+bias] [+resid].
// 16 rows per block, 8 waves x 2 column tiles. A tile staged to LDS once and
// held in registers (8 fragments); B fragments stream straight from the
// [col][k] bf16 weight (L2/WGP$-resident). No barriers in the K loop.
// ---------------------------------------------------------------------------
template<bool OUT_BF16>
__global__ __launch_bounds__(256)
void gemm256t(const float* __restrict__ A, const __bf16* __restrict__ Wt,
              const float* __restrict__ bias, const float* __restrict__ resid,
              void* __restrict__ outp, float alpha)
{
  const int rowBase = blockIdx.x * 16;
  const int tid  = threadIdx.x;
  const int lane = tid & 31;
  const int wave = tid >> 5;

  __shared__ __align__(16) __bf16 sA[16 * 264];   // full 16x256 A tile (bf16)

  // stage full A tile once: 4096 elems, 16 per thread (coalesced along k)
  {
    const int r = tid >> 4;                       // 16 threads per row
    const int ks = (tid & 15) * 16;               // 16 contiguous k each
    const float* ap = A + (size_t)(rowBase + r) * DMODEL + ks;
    #pragma unroll
    for (int j = 0; j < 16; ++j) sA[r * 264 + ks + j] = (__bf16)ap[j];
  }
  __syncthreads();

  v16bf afrag[8];
  #pragma unroll
  for (int kc = 0; kc < 8; ++kc) afrag[kc] = load_frag(sA + kc * 32, 264, lane);

  v8f acc[2] = { v8f{}, v8f{} };
  #pragma unroll
  for (int t = 0; t < 2; ++t) {
    const int colBase = ((t ? wave + 8 : wave) * 16);
    const __bf16* wp = Wt + (size_t)colBase * DMODEL;
    v16bf bfrag[8];                               // 16-load clause
    #pragma unroll
    for (int kc = 0; kc < 8; ++kc) bfrag[kc] = load_frag(wp + kc * 32, DMODEL, lane);
    #pragma unroll
    for (int kc = 0; kc < 8; ++kc)                // 8 back-to-back WMMAs
      acc[t] = __builtin_amdgcn_wmma_f32_16x16x32_bf16(false, afrag[kc], false, bfrag[kc],
                                                       (short)0, acc[t], false, false);
  }

  const int colLane = lane & 15;
  const int rsub = (lane >> 4) * 8;
  #pragma unroll
  for (int t = 0; t < 2; ++t) {
    const int col = ((t ? wave + 8 : wave) * 16) + colLane;
    #pragma unroll
    for (int r = 0; r < 8; ++r) {
      const int row = rowBase + rsub + r;
      float v = alpha * acc[t][r];
      if (bias)  v += bias[col];
      if (resid) v += resid[(size_t)row * DMODEL + col];
      if (OUT_BF16) ((__bf16*)outp)[(size_t)row * DMODEL + col] = (__bf16)v;
      else          ((float *)outp)[(size_t)row * DMODEL + col] = v;
    }
  }
}

// ---------------------------------------------------------------------------
// Fused scores (Q@K^T, pre-scaled) + streaming top-32 + softmax.
// Q fragments live in registers for the whole kernel; per 128-cell step each
// wave issues one 16-load clause for K then 8 back-to-back WMMAs.
// ---------------------------------------------------------------------------
__global__ __launch_bounds__(256)
void scores_topk(const __bf16* __restrict__ Q, const __bf16* __restrict__ K,
                 float* __restrict__ attnOut, int* __restrict__ idxOut)
{
  const int b     = blockIdx.x >> 7;              // 128 query tiles / batch
  const int qBase = (blockIdx.x & 127) * 16;
  const int tid  = threadIdx.x;
  const int lane = tid & 31;
  const int wave = tid >> 5;

  __shared__ __align__(16) __bf16 sQ[16 * 264];   // 16x256 Q tile
  __shared__ float sS[16 * 132];                  // 16x128 score tile
  __shared__ float tV[16][TOPKN];
  __shared__ int   tI[16][TOPKN];
  __shared__ float thr[16];
  __shared__ int   cnt[16];
  __shared__ int   fill[16];
  __shared__ float cV[16][128];
  __shared__ int   cI[16][128];

  if (tid < 16) { thr[tid] = -3.0e38f; cnt[tid] = 0; fill[tid] = 0; }
  for (int e = tid; e < 16 * TOPKN; e += 256) {
    tV[e >> 5][e & 31] = -3.0e38f; tI[e >> 5][e & 31] = 0;
  }
  for (int c = tid; c < 16 * 32; c += 256) {      // vectorized Q tile load
    int r = c >> 5, koff = (c & 31) * 8;
    *(uint4*)(sQ + r * 264 + koff) =
        *(const uint4*)(Q + (size_t)(b * ND + qBase + r) * DMODEL + koff);
  }
  __syncthreads();

  // hoist all 8 A fragments into registers; no DS traffic in the hot loop
  v16bf afrag[8];
  #pragma unroll
  for (int kc = 0; kc < 8; ++kc) afrag[kc] = load_frag(sQ + kc * 32, 264, lane);

  const __bf16* Kb = K + (size_t)b * NC * DMODEL;

  for (int m0 = 0; m0 < NC; m0 += 128) {
    const int c0 = m0 + wave * 16;
    const __bf16* kp = Kb + (size_t)c0 * DMODEL;

    v16bf bfrag[8];                               // 16-load clause
    #pragma unroll
    for (int kc = 0; kc < 8; ++kc) bfrag[kc] = load_frag(kp + kc * 32, DMODEL, lane);

    v8f acc = {};
    #pragma unroll
    for (int kc = 0; kc < 8; ++kc)                // 8 back-to-back WMMAs
      acc = __builtin_amdgcn_wmma_f32_16x16x32_bf16(false, afrag[kc], false, bfrag[kc],
                                                    (short)0, acc, false, false);
    {
      const int cl = wave * 16 + (lane & 15);
      const int q0 = (lane >> 4) * 8;
      #pragma unroll
      for (int r = 0; r < 8; ++r) sS[(q0 + r) * 132 + cl] = acc[r];
    }
    __syncthreads();

    {   // filter: 16 threads per query against running threshold
      const int q = tid >> 4, i = tid & 15;
      const float th = thr[q];
      #pragma unroll
      for (int j = 0; j < 8; ++j) {
        int c = i * 8 + j;
        float v = sS[q * 132 + c];
        if (v > th) {
          int pos = atomicAdd(&cnt[q], 1);
          cV[q][pos] = v; cI[q][pos] = m0 + c;
        }
      }
    }
    __syncthreads();

    if (tid < 16) {                               // owner-thread sorted insert
      const int q = tid;
      int f = fill[q];
      const int n = cnt[q];
      for (int s = 0; s < n; ++s) {
        float v = cV[q][s]; int id = cI[q][s];
        if (f == TOPKN && v <= tV[q][TOPKN - 1]) continue;
        int pos = (f < TOPKN) ? f : (TOPKN - 1);
        while (pos > 0 && tV[q][pos - 1] < v) {
          tV[q][pos] = tV[q][pos - 1]; tI[q][pos] = tI[q][pos - 1]; --pos;
        }
        tV[q][pos] = v; tI[q][pos] = id;
        if (f < TOPKN) ++f;
      }
      fill[q] = f; cnt[q] = 0;
      thr[q] = (f == TOPKN) ? tV[q][TOPKN - 1] : -3.0e38f;
    }
    __syncthreads();
  }

  if (tid < 16) {                                 // softmax (list desc-sorted)
    const int q = tid;
    const float mx = tV[q][0];
    float ex[TOPKN], sum = 0.f;
    #pragma unroll
    for (int k = 0; k < TOPKN; ++k) { ex[k] = __expf(tV[q][k] - mx); sum += ex[k]; }
    const float inv = 1.f / sum;
    const size_t o = (size_t)(b * ND + qBase + q) * TOPKN;
    #pragma unroll
    for (int k = 0; k < TOPKN; ++k) { attnOut[o + k] = ex[k] * inv; idxOut[o + k] = tI[q][k]; }
  }
}

// ---------------------------------------------------------------------------
// context[n,:] = sum_k attn[n,k] * V_full[b, idx[n,k], :]   (V is L2-resident)
// ---------------------------------------------------------------------------
__global__ __launch_bounds__(256)
void context_gather(const float* __restrict__ attn, const int* __restrict__ idx,
                    const float* __restrict__ V, float* __restrict__ ctx)
{
  const int rowBase = blockIdx.x * 16;
  const int b = rowBase / ND;
  const int tid = threadIdx.x;                    // column
  __shared__ float sw[16][TOPKN];
  __shared__ int   si[16][TOPKN];
  for (int e = tid; e < 16 * TOPKN; e += 256) {
    int r = e >> 5, k = e & 31;
    sw[r][k] = attn[(size_t)(rowBase + r) * TOPKN + k];
    si[r][k] = idx [(size_t)(rowBase + r) * TOPKN + k];
  }
  __syncthreads();
  const float* Vb = V + (size_t)b * NC * DMODEL;
  for (int r = 0; r < 16; ++r) {
    float acc = 0.f;
    #pragma unroll 8
    for (int k = 0; k < TOPKN; ++k)
      acc += sw[r][k] * Vb[(size_t)si[r][k] * DMODEL + tid];
    ctx[(size_t)(rowBase + r) * DMODEL + tid] = acc;
  }
}

// ---------------------------------------------------------------------------
extern "C" void kernel_launch(void* const* d_in, const int* in_sizes, int n_in,
                              void* d_out, int out_size, void* d_ws, size_t ws_size,
                              hipStream_t stream)
{
  const float* micro = (const float*)d_in[0];
  const float* macro = (const float*)d_in[1];
  const float* mp_w  = (const float*)d_in[2];
  const float* mp_b  = (const float*)d_in[3];
  const float* wq    = (const float*)d_in[4];
  const float* wk    = (const float*)d_in[5];
  const float* wv    = (const float*)d_in[6];
  const float* op_w  = (const float*)d_in[7];
  const float* op_b  = (const float*)d_in[8];

  char* w = (char*)d_ws;
  float*  xm  = (float*) w; w += (size_t)B_ * ND * DMODEL * 4;   // x_micro f32
  __bf16* Qb  = (__bf16*)w; w += (size_t)B_ * ND * DMODEL * 2;   // Q (scaled) bf16
  __bf16* Kb  = (__bf16*)w; w += (size_t)B_ * NC * DMODEL * 2;   // K_full bf16
  float*  Vf  = (float*) w; w += (size_t)B_ * NC * DMODEL * 4;   // V_full f32
  float*  ctx = (float*) w; w += (size_t)B_ * ND * DMODEL * 4;   // context f32
  __bf16* wqT = (__bf16*)w; w += (size_t)DMODEL * DMODEL * 2;    // [col][k] bf16
  __bf16* wkT = (__bf16*)w; w += (size_t)DMODEL * DMODEL * 2;
  __bf16* wvT = (__bf16*)w; w += (size_t)DMODEL * DMODEL * 2;
  __bf16* opT = (__bf16*)w; w += (size_t)DMODEL * DMODEL * 2;

  float* xcond = (float*)d_out;                              // [B,Nd,D]
  float* attn  = xcond + (size_t)B_ * ND * DMODEL;           // [B,Nd,32]
  int*   idx   = (int*)(attn + (size_t)B_ * ND * TOPKN);     // [B,Nd,32] int32 bits

  const float scale = 0.0625f;                               // 256^-0.5

  transpose_w   <<<64, 256, 0, stream>>>(wq,   wqT);
  transpose_w   <<<64, 256, 0, stream>>>(wk,   wkT);
  transpose_w   <<<64, 256, 0, stream>>>(wv,   wvT);
  transpose_w   <<<64, 256, 0, stream>>>(op_w, opT);
  micro_proj    <<<B_ * ND / 16, 256, 0, stream>>>(micro, mp_w, mp_b, xm);
  gemm256t<true ><<<B_ * ND / 16, 256, 0, stream>>>(xm,    wqT, nullptr, nullptr, Qb, scale);
  gemm256t<true ><<<B_ * NC / 16, 256, 0, stream>>>(macro, wkT, nullptr, nullptr, Kb, 1.0f);
  gemm256t<false><<<B_ * NC / 16, 256, 0, stream>>>(macro, wvT, nullptr, nullptr, Vf, 1.0f);
  scores_topk   <<<B_ * ND / 16, 256, 0, stream>>>(Qb, Kb, attn, idx);
  context_gather<<<B_ * ND / 16, 256, 0, stream>>>(attn, idx, Vf, ctx);
  gemm256t<false><<<B_ * ND / 16, 256, 0, stream>>>(ctx, opT, op_b, xm, xcond, 1.0f);
}
